// SparseMoE_Self_Attention_48052094107926
// MI455X (gfx1250) — compile-verified, matrix operand order checked by reading, not run on hardware
//
#include <hip/hip_runtime.h>

#define B_ROWS 8192
#define DIM    2048
#define NEXP   8

typedef __attribute__((ext_vector_type(16))) __bf16 v16bf;
typedef __attribute__((ext_vector_type(8)))  float  v8f;
typedef __attribute__((ext_vector_type(2)))  float  v2f;
typedef __attribute__((ext_vector_type(4)))  float  v4f;
typedef __attribute__((ext_vector_type(4)))  __bf16 v4bf;

// ---------------------------------------------------------------------------
// x (f32) -> xb (bf16), vectorized 4-wide
// ---------------------------------------------------------------------------
__global__ __launch_bounds__(256) void convert_x_bf16(const float* __restrict__ x,
                                                      __bf16* __restrict__ xb, int n4) {
  int i = blockIdx.x * 256 + threadIdx.x;
  if (i < n4) {
    v4f f = ((const v4f*)x)[i];
    ((v4bf*)xb)[i] = __builtin_convertvector(f, v4bf);
  }
}

// ---------------------------------------------------------------------------
// dst[e][m][d] = bf16( sum_{t<nsum} src[e][d][t*sumStride + m] )
// LDS-tiled transpose so both global read and write are coalesced.
// Used for Wsum (nsum=3, folds q+k+v) and W_proj (nsum=1).
// ---------------------------------------------------------------------------
__global__ __launch_bounds__(256) void transpose_sum_bf16(
    const float* __restrict__ src, __bf16* __restrict__ dst,
    int nsum, int srcRowStride, int sumStride,
    long srcBatchStride, long dstBatchStride) {
  __shared__ float tile[32][33];
  int e  = blockIdx.z;
  int d0 = blockIdx.x * 32, m0 = blockIdx.y * 32;
  int tx = threadIdx.x, ty = threadIdx.y;
  const float* s = src + (long)e * srcBatchStride;
  #pragma unroll
  for (int dy = 0; dy < 32; dy += 8) {
    int d = d0 + ty + dy, m = m0 + tx;
    float v = 0.f;
    for (int t = 0; t < nsum; ++t) v += s[(long)d * srcRowStride + t * sumStride + m];
    tile[ty + dy][tx] = v;
  }
  __syncthreads();
  __bf16* o = dst + (long)e * dstBatchStride;
  #pragma unroll
  for (int dy = 0; dy < 32; dy += 8) {
    int m = m0 + ty + dy, d = d0 + tx;
    o[(long)m * DIM + d] = (__bf16)tile[tx][ty + dy];
  }
}

// ---------------------------------------------------------------------------
// Gating: logits = x @ W_gate + b_gate, softmax, top-2.
// Writes sparse gate weights gw (B x 8) and pushes (row<<1)|slot into the
// per-expert routing list. Full f32 (it controls routing -> keep exact).
// ---------------------------------------------------------------------------
__global__ __launch_bounds__(256) void gate_topk(
    const float* __restrict__ x, const float* __restrict__ wg,
    const float* __restrict__ bg, float* __restrict__ gw,
    int* __restrict__ counts, int* __restrict__ rowlist) {
  int row = blockIdx.x, tid = threadIdx.x;
  float acc[NEXP];
  #pragma unroll
  for (int e = 0; e < NEXP; ++e) acc[e] = 0.f;
  const float* xr = x + (size_t)row * DIM;
  for (int d = tid; d < DIM; d += 256) {
    float xv = xr[d];
    #pragma unroll
    for (int e = 0; e < NEXP; ++e) acc[e] += xv * wg[d * NEXP + e];
  }
  __shared__ float red[256];
  __shared__ float logits[NEXP];
  for (int e = 0; e < NEXP; ++e) {
    red[tid] = acc[e];
    __syncthreads();
    for (int s2 = 128; s2 > 0; s2 >>= 1) {
      if (tid < s2) red[tid] += red[tid + s2];
      __syncthreads();
    }
    if (tid == 0) logits[e] = red[0] + bg[e];
    __syncthreads();
  }
  if (tid == 0) {
    float mx = logits[0];
    #pragma unroll
    for (int e = 1; e < NEXP; ++e) mx = fmaxf(mx, logits[e]);
    float p[NEXP], se = 0.f;
    #pragma unroll
    for (int e = 0; e < NEXP; ++e) { p[e] = __expf(logits[e] - mx); se += p[e]; }
    float inv = 1.f / se;
    #pragma unroll
    for (int e = 0; e < NEXP; ++e) p[e] *= inv;
    int i1 = 0; float v1 = p[0];
    #pragma unroll
    for (int e = 1; e < NEXP; ++e) if (p[e] > v1) { v1 = p[e]; i1 = e; }
    int i2 = -1; float v2 = -1.f;
    #pragma unroll
    for (int e = 0; e < NEXP; ++e) if (e != i1 && p[e] > v2) { v2 = p[e]; i2 = e; }
    #pragma unroll
    for (int e = 0; e < NEXP; ++e)
      gw[row * NEXP + e] = (e == i1) ? v1 : ((e == i2) ? v2 : 0.f);
    int pos = atomicAdd(&counts[i1], 1);
    rowlist[i1 * B_ROWS + pos] = (row << 1);
    pos = atomicAdd(&counts[i2], 1);
    rowlist[i2 * B_ROWS + pos] = (row << 1) | 1;
  }
}

// ---------------------------------------------------------------------------
// Routed MoE GEMM: s_part[slot][row] = gw[row,e] * (x[row] @ Wsum[e])
// bf16 WMMA 16x16x32, f32 accumulate.
// Block = 8 waves, block tile 64x128; each wave owns a 32x32 tile
// (2x2 accumulators) so every A/B fragment feeds two WMMAs: 1 load/wmma.
// Each row belongs to exactly one list per slot -> plain stores, no atomics.
// ---------------------------------------------------------------------------
__global__ __launch_bounds__(256) void moe_gemm_bf16(
    const __bf16* __restrict__ xb, const __bf16* __restrict__ wsumT,
    const float* __restrict__ gw, const int* __restrict__ counts,
    const int* __restrict__ rowlist, float* __restrict__ spart) {
  int e = blockIdx.z;
  int cnt = counts[e];
  int rowBase = blockIdx.y * 64;
  if (rowBase >= cnt) return;          // early-exit for over-provisioned grid
  __shared__ int   rrow[64];
  __shared__ int   rslot[64];
  __shared__ float rw[64];
  int tid = threadIdx.x;
  if (tid < 64) {
    int idx = rowBase + tid;
    bool v = idx < cnt;
    int entry = rowlist[e * B_ROWS + (v ? idx : rowBase)];
    rrow[tid]  = entry >> 1;
    rslot[tid] = v ? (entry & 1) : -1;
    rw[tid]    = v ? gw[(entry >> 1) * NEXP + e] : 0.f;
  }
  __syncthreads();
  int wave = tid >> 5, lane = tid & 31;
  int waveM = wave >> 2, waveN = wave & 3;      // 2 x 4 wave grid
  int l16 = lane & 15, half = lane >> 4;
  int mBase = waveM * 32;                       // within block tile
  int colBase = blockIdx.x * 128 + waveN * 32;
  // A fragments: lane holds 16 contiguous bf16 of its row, K-half by lane/16
  const __bf16* aptr0 = xb + (size_t)rrow[mBase + l16] * DIM + half * 16;
  const __bf16* aptr1 = xb + (size_t)rrow[mBase + 16 + l16] * DIM + half * 16;
  // B fragments: WsumT is N-major, so lane's 16 K-values are contiguous too
  const __bf16* bptr0 = wsumT + ((size_t)e * DIM + colBase + l16) * DIM + half * 16;
  const __bf16* bptr1 = bptr0 + (size_t)16 * DIM;
  v8f c00 = {}, c01 = {}, c10 = {}, c11 = {};
  #pragma unroll 2
  for (int k0 = 0; k0 < DIM; k0 += 32) {
    v16bf a0 = *(const v16bf*)(aptr0 + k0);
    v16bf a1 = *(const v16bf*)(aptr1 + k0);
    v16bf b0 = *(const v16bf*)(bptr0 + k0);
    v16bf b1 = *(const v16bf*)(bptr1 + k0);
    c00 = __builtin_amdgcn_wmma_f32_16x16x32_bf16(false, a0, false, b0, (short)0, c00, false, false);
    c01 = __builtin_amdgcn_wmma_f32_16x16x32_bf16(false, a0, false, b1, (short)0, c01, false, false);
    c10 = __builtin_amdgcn_wmma_f32_16x16x32_bf16(false, a1, false, b0, (short)0, c10, false, false);
    c11 = __builtin_amdgcn_wmma_f32_16x16x32_bf16(false, a1, false, b1, (short)0, c11, false, false);
  }
  #pragma unroll
  for (int mi = 0; mi < 2; ++mi) {
    v8f cv0 = mi ? c10 : c00;
    v8f cv1 = mi ? c11 : c01;
    #pragma unroll
    for (int r = 0; r < 8; ++r) {
      int mloc = mBase + mi * 16 + r + 8 * half;  // C layout: VGPR r -> M=r (+8 hi half)
      int slot = rslot[mloc];
      if (slot >= 0) {
        size_t dst = (size_t)slot * B_ROWS * DIM + (size_t)rrow[mloc] * DIM;
        float w = rw[mloc];
        spart[dst + colBase + l16]      = cv0[r] * w;
        spart[dst + colBase + 16 + l16] = cv1[r] * w;
      }
    }
  }
}

// ---------------------------------------------------------------------------
// Per-row head attention, one wave per row, full f32 via WMMA 16x16x4.
//   S = s0[b]+s1[b]  (16x128);  A = softmax(S S^T * hd^-0.5);  O = A S
// Output written bf16, already in the transpose(0,2,1) interleave (j*16+h),
// ready to be the A operand of the projection GEMM.
// ---------------------------------------------------------------------------
__global__ __launch_bounds__(128) void attn_rows(const float* __restrict__ spart,
                                                 __bf16* __restrict__ outb) {
  __shared__ float Sl[4][16 * 130];   // row stride 130 to dodge bank conflicts
  __shared__ float Al[4][16 * 17];
  int wave = threadIdx.x >> 5, lane = threadIdx.x & 31;
  int b = blockIdx.x * 4 + wave;
  float* Sw = &Sl[wave][0];
  float* Aw = &Al[wave][0];
  const float* s0 = spart + (size_t)b * DIM;
  const float* s1 = spart + (size_t)B_ROWS * DIM + (size_t)b * DIM;
  for (int i = lane; i < DIM; i += 32)
    Sw[(i >> 7) * 130 + (i & 127)] = s0[i] + s1[i];
  __syncthreads();
  int l16 = lane & 15, half = lane >> 4;
  // --- S @ S^T : M=N=16, K=128 -> 32x wmma_f32_16x16x4. A and B fragments coincide.
  v8f c = {};
  #pragma unroll
  for (int kc = 0; kc < 32; ++kc) {
    int k = kc * 4 + half * 2;
    v2f a;
    a.x = Sw[l16 * 130 + k];
    a.y = Sw[l16 * 130 + k + 1];
    c = __builtin_amdgcn_wmma_f32_16x16x4_f32(false, a, false, a, (short)0, c, false, false);
  }
  const float scale = 0.088388347648318447f;  // 128^-0.5
  // --- row softmax: row lives in a VGPR, columns across 16 lanes -> shfl_xor<16
  #pragma unroll
  for (int r = 0; r < 8; ++r) {
    float v = c[r] * scale;
    float m = v;
    #pragma unroll
    for (int off = 1; off < 16; off <<= 1) m = fmaxf(m, __shfl_xor(m, off, 32));
    float ev = __expf(v - m);
    float sm = ev;
    #pragma unroll
    for (int off = 1; off < 16; off <<= 1) sm += __shfl_xor(sm, off, 32);
    float p = ev / sm;
    Aw[(r + 8 * half) * 17 + l16] = p;      // stage attn for transpose to A-layout
  }
  __syncthreads();
  // --- O = A(16x16) @ S(16x128): per 16-col chunk, K=16 -> 4x wmma_f32_16x16x4
  #pragma unroll
  for (int nc = 0; nc < 8; ++nc) {
    v8f c2 = {};
    #pragma unroll
    for (int kc = 0; kc < 4; ++kc) {
      int kb = kc * 4 + half * 2;
      v2f a2, b2;
      a2.x = Aw[l16 * 17 + kb];
      a2.y = Aw[l16 * 17 + kb + 1];
      b2.x = Sw[kb * 130 + nc * 16 + l16];
      b2.y = Sw[(kb + 1) * 130 + nc * 16 + l16];
      c2 = __builtin_amdgcn_wmma_f32_16x16x4_f32(false, a2, false, b2, (short)0, c2, false, false);
    }
    #pragma unroll
    for (int r = 0; r < 8; ++r) {
      int h = r + 8 * half;
      int j = nc * 16 + l16;
      outb[(size_t)b * DIM + j * 16 + h] = (__bf16)c2[r];   // transpose(0,2,1) interleave
    }
  }
}

// ---------------------------------------------------------------------------
// Final projection: out = outb @ W_proj + b_proj.  bf16 WMMA, f32 out.
// Same 32x32-per-wave register blocking as the MoE GEMM.
// ---------------------------------------------------------------------------
__global__ __launch_bounds__(256) void proj_gemm_bf16(
    const __bf16* __restrict__ ab, const __bf16* __restrict__ wprojT,
    const float* __restrict__ bias, float* __restrict__ out) {
  int tid = threadIdx.x;
  int wave = tid >> 5, lane = tid & 31;
  int waveM = wave >> 2, waveN = wave & 3;
  int l16 = lane & 15, half = lane >> 4;
  int row0 = blockIdx.y * 64 + waveM * 32;
  int colBase = blockIdx.x * 128 + waveN * 32;
  const __bf16* aptr0 = ab + (size_t)(row0 + l16) * DIM + half * 16;
  const __bf16* aptr1 = aptr0 + (size_t)16 * DIM;
  const __bf16* bptr0 = wprojT + (size_t)(colBase + l16) * DIM + half * 16;
  const __bf16* bptr1 = bptr0 + (size_t)16 * DIM;
  v8f c00 = {}, c01 = {}, c10 = {}, c11 = {};
  #pragma unroll 2
  for (int k0 = 0; k0 < DIM; k0 += 32) {
    v16bf a0 = *(const v16bf*)(aptr0 + k0);
    v16bf a1 = *(const v16bf*)(aptr1 + k0);
    v16bf b0 = *(const v16bf*)(bptr0 + k0);
    v16bf b1 = *(const v16bf*)(bptr1 + k0);
    c00 = __builtin_amdgcn_wmma_f32_16x16x32_bf16(false, a0, false, b0, (short)0, c00, false, false);
    c01 = __builtin_amdgcn_wmma_f32_16x16x32_bf16(false, a0, false, b1, (short)0, c01, false, false);
    c10 = __builtin_amdgcn_wmma_f32_16x16x32_bf16(false, a1, false, b0, (short)0, c10, false, false);
    c11 = __builtin_amdgcn_wmma_f32_16x16x32_bf16(false, a1, false, b1, (short)0, c11, false, false);
  }
  float bs0 = bias[colBase + l16];
  float bs1 = bias[colBase + 16 + l16];
  #pragma unroll
  for (int mi = 0; mi < 2; ++mi) {
    v8f cv0 = mi ? c10 : c00;
    v8f cv1 = mi ? c11 : c01;
    #pragma unroll
    for (int r = 0; r < 8; ++r) {
      size_t row = row0 + mi * 16 + r + 8 * half;
      out[row * DIM + colBase + l16]      = cv0[r] + bs0;
      out[row * DIM + colBase + 16 + l16] = cv1[r] + bs1;
    }
  }
}

// ---------------------------------------------------------------------------
extern "C" void kernel_launch(void* const* d_in, const int* in_sizes, int n_in,
                              void* d_out, int out_size, void* d_ws, size_t ws_size,
                              hipStream_t stream) {
  const float* x      = (const float*)d_in[0];
  const float* W_qkv  = (const float*)d_in[1];
  const float* W_gate = (const float*)d_in[2];
  const float* b_gate = (const float*)d_in[3];
  const float* W_proj = (const float*)d_in[4];
  const float* b_proj = (const float*)d_in[5];
  float* out = (float*)d_out;

  // workspace carve-out (256B aligned)
  char* base = (char*)d_ws;
  size_t off = 0;
  auto alloc = [&](size_t bytes) -> char* {
    char* p = base + off;
    off = (off + bytes + 255) & ~(size_t)255;
    return p;
  };
  __bf16* xb      = (__bf16*)alloc((size_t)B_ROWS * DIM * 2);
  __bf16* wsumT   = (__bf16*)alloc((size_t)NEXP * DIM * DIM * 2);
  __bf16* wprojT  = (__bf16*)alloc((size_t)DIM * DIM * 2);
  float*  gw      = (float*) alloc((size_t)B_ROWS * NEXP * 4);
  int*    counts  = (int*)   alloc(NEXP * 4);
  int*    rowlist = (int*)   alloc((size_t)NEXP * B_ROWS * 4);
  float*  spart   = (float*) alloc((size_t)2 * B_ROWS * DIM * 4);
  __bf16* outb    = (__bf16*)alloc((size_t)B_ROWS * DIM * 2);

  hipMemsetAsync(counts, 0, NEXP * sizeof(int), stream);

  // x -> bf16
  int n4 = B_ROWS * DIM / 4;
  convert_x_bf16<<<n4 / 256, 256, 0, stream>>>(x, xb, n4);

  // Wsum[e] = sum of q/k/v column blocks, transposed to N-major bf16
  transpose_sum_bf16<<<dim3(DIM / 32, DIM / 32, NEXP), dim3(32, 8), 0, stream>>>(
      W_qkv, wsumT, 3, 3 * DIM, DIM, (long)DIM * 3 * DIM, (long)DIM * DIM);
  // W_proj transposed to N-major bf16
  transpose_sum_bf16<<<dim3(DIM / 32, DIM / 32, 1), dim3(32, 8), 0, stream>>>(
      W_proj, wprojT, 1, DIM, 0, 0, 0);

  // gating + top-2 routing
  gate_topk<<<B_ROWS, 256, 0, stream>>>(x, W_gate, b_gate, gw, counts, rowlist);

  // routed MoE GEMM (grid over-provisioned; blocks past count[e] exit early)
  moe_gemm_bf16<<<dim3(DIM / 128, B_ROWS / 64, NEXP), 256, 0, stream>>>(
      xb, wsumT, gw, counts, rowlist, spart);

  // per-row attention (f32 WMMA), emits bf16 interleaved output
  attn_rows<<<B_ROWS / 4, 128, 0, stream>>>(spart, outb);

  // final projection + bias
  proj_gemm_bf16<<<dim3(DIM / 128, B_ROWS / 64), 256, 0, stream>>>(outb, wprojT, b_proj, out);
}
